// FGDA_38895223832844
// MI455X (gfx1250) — matrix operand order repaired and checked by reading, not compiled, not run on hardware
//
#include <hip/hip_runtime.h>
#include <math.h>

// ---------- compile-time problem shape ----------
constexpr int NB = 8, HH = 128, WW = 128;     // batch, height, width
constexpr int PTOT = NB * HH * WW;            // 131072 pixels

// ---------- types ----------
typedef __bf16 bf16_t;
typedef __attribute__((ext_vector_type(16))) __bf16 v16bf;
typedef __attribute__((ext_vector_type(8)))  float  v8f;
typedef __attribute__((ext_vector_type(4)))  unsigned int v4u;
typedef __attribute__((ext_vector_type(4)))  int v4i;
typedef __attribute__((ext_vector_type(8)))  int v8i;

__device__ __forceinline__ int iclamp(int v, int lo, int hi) {
    return v < lo ? lo : (v > hi ? hi : v);
}
__device__ __forceinline__ float lrelu_f(float v) { return v >= 0.f ? v : 0.1f * v; }

// ---------------------------------------------------------------------------
// TDM: 1-D DMA of `bytes` (multiple of 8) from global -> LDS.
// Descriptor per CDNA5 ISA ch.8: group0 {count=1, lds_addr, global_addr, type=2},
// group1 {data_size=8B, tensor_dim0=tile_dim0=stride=bytes/8, tile_dim1=1}.
// ---------------------------------------------------------------------------
__device__ __forceinline__ void tdm_load_1d(const void* gsrc, unsigned lds_off, int bytes) {
    unsigned long long ga = (unsigned long long)(size_t)gsrc;
    unsigned d0 = (unsigned)(bytes >> 3);   // 8-byte units, fits 16 bits
    v4u g0 = { 1u,                                    // count=1 (valid descriptor)
               lds_off,                               // lds_addr
               (unsigned)(ga & 0xFFFFFFFFu),          // global_addr[31:0]
               (unsigned)((ga >> 32) & 0x1FFFFFFu) | (2u << 30) };  // addr[56:32] | type=2
    v8i g1 = { (int)(3u << 16),                       // data_size = 3 (8 bytes)
               (int)(d0 << 16),                       // tensor_dim0[15:0]
               (int)((d0 >> 16) | (1u << 16)),        // tensor_dim0 hi | tensor_dim1=1
               (int)(d0 << 16),                       // tile_dim0
               1,                                     // tile_dim1 = 1, tile_dim2 = 0
               (int)d0,                               // tensor_dim0_stride lo
               0, 0 };
    v4i z4 = {0, 0, 0, 0};
#if defined(__clang_major__) && __clang_major__ >= 23
    v8i z8 = {0, 0, 0, 0, 0, 0, 0, 0};
    __builtin_amdgcn_tensor_load_to_lds(g0, g1, z4, z4, z8, 0);
#else
    __builtin_amdgcn_tensor_load_to_lds(g0, g1, z4, z4, 0);
#endif
}

// ---------------------------------------------------------------------------
// Weight repack: OIHW fp32 -> bf16 A-fragment-friendly [cos][K/32][64][32]
// k = (ky*3+kx)*Cin + c  (matches NHWC activation / sampled-tensor K order)
// ---------------------------------------------------------------------------
__global__ void pack_weights_kernel(const float* __restrict__ w, bf16_t* __restrict__ wpk,
                                    int Cout, int Cin, int taps, int CoutPad) {
    unsigned total = (unsigned)CoutPad * Cin * taps;
    unsigned tid = blockIdx.x * blockDim.x + threadIdx.x;
    if (tid >= total) return;
    int kl = (int)(tid & 31u);
    unsigned t = tid >> 5;
    int mr = (int)(t & 63u);
    t >>= 6;                                  // t = cos*KC + kc
    int KC = (Cin * taps) >> 5;
    int kc  = (int)(t % (unsigned)KC);
    int cos = (int)(t / (unsigned)KC);
    int co  = cos * 64 + mr;
    int k   = kc * 32 + kl;
    int tap = k / Cin, c = k % Cin;
    float v = 0.f;
    if (co < Cout) v = w[(co * Cin + c) * taps + tap];
    wpk[tid] = (bf16_t)v;
}

// ---------------------------------------------------------------------------
// backwarp(nbr, flow) and concat with ref -> x0 NHWC bf16 [P][128]
// ---------------------------------------------------------------------------
__global__ void warp_concat_kernel(const float* __restrict__ nbr, const float* __restrict__ ref,
                                   const float* __restrict__ flow, bf16_t* __restrict__ x0) {
    unsigned tid = blockIdx.x * blockDim.x + threadIdx.x;   // < PTOT*64
    int c = (int)(tid & 63u);
    int p = (int)(tid >> 6);
    int w = p & (WW - 1), h = (p >> 7) & (HH - 1), n = p >> 14;
    constexpr int HW = HH * WW;
    int pix = h * WW + w;
    float px = (float)w + flow[(n * 2 + 0) * HW + pix];
    float py = (float)h + flow[(n * 2 + 1) * HW + pix];
    float y0f = floorf(py), x0f = floorf(px);
    float dy = py - y0f, dx = px - x0f;
    int y0 = (int)y0f, x0i = (int)x0f, y1 = y0 + 1, x1 = x0i + 1;
    float w00 = (1.f - dy) * (1.f - dx) * ((y0 >= 0 && y0 < HH && x0i >= 0 && x0i < WW) ? 1.f : 0.f);
    float w01 = (1.f - dy) * dx         * ((y0 >= 0 && y0 < HH && x1  >= 0 && x1  < WW) ? 1.f : 0.f);
    float w10 = dy * (1.f - dx)         * ((y1 >= 0 && y1 < HH && x0i >= 0 && x0i < WW) ? 1.f : 0.f);
    float w11 = dy * dx                 * ((y1 >= 0 && y1 < HH && x1  >= 0 && x1  < WW) ? 1.f : 0.f);
    int y0c = iclamp(y0, 0, HH - 1), y1c = iclamp(y1, 0, HH - 1);
    int x0c = iclamp(x0i, 0, WW - 1), x1c = iclamp(x1, 0, WW - 1);
    const float* base = nbr + (n * 64 + c) * HW;
    float v = base[y0c * WW + x0c] * w00 + base[y0c * WW + x1c] * w01 +
              base[y1c * WW + x0c] * w10 + base[y1c * WW + x1c] * w11;
    x0[(size_t)p * 128 + c]      = (bf16_t)v;
    x0[(size_t)p * 128 + 64 + c] = (bf16_t)ref[(n * 64 + c) * HW + pix];
}

// ---------------------------------------------------------------------------
// Generic implicit-GEMM conv (3x3 or 1x1) on bf16 NHWC with WMMA bf16.
//   block = 256 threads = 8 waves = (4 m-tiles of 16 out-ch) x (2 n-tiles),
//   each wave owns TWO 16-px N-subtiles -> block tile = 64 out-ch x 64 px,
//   so every LDS A-fragment feeds 2 WMMAs.
//   Weights stream through a DOUBLE-BUFFERED 2x24KB LDS stage driven by the
//   Tensor Data Mover (issue kb+1, compute kb, s_wait_tensorcnt, barrier).
// OUTF=0: bf16 NHWC [P][COUT]; OUTF=1: fp32 NCHW (final output).
// ---------------------------------------------------------------------------
template <int CIN, int TAPS, int COUT, bool RELU, int OUTF>
__global__ __launch_bounds__(256) void conv_wmma_kernel(
        const bf16_t* __restrict__ act, const bf16_t* __restrict__ wpk,
        const float* __restrict__ bias, void* __restrict__ outp) {
    constexpr int KTOT = CIN * TAPS;
    constexpr int KB   = 192;          // K per LDS stage (divides 576 and 1152)
    constexpr int NKB  = KTOT / KB;
    constexpr int KCB  = KB / 32;      // 6 WMMA K-steps per stage

    __shared__ bf16_t wlds[2][64 * KB];   // double-buffered 24 KB weight tiles

    const int tid = threadIdx.x;
    const int wave = tid >> 5, lane = tid & 31;
    const int mt = wave & 3, nt = wave >> 1 >> 1;   // mt 0..3, nt 0..1
    const int l16 = lane & 15, lh = lane >> 4;

    const int pbase = blockIdx.x * 64 + nt * 32;
    int pp[2], pw[2], ph[2], pn[2];
#pragma unroll
    for (int s = 0; s < 2; ++s) {
        pp[s] = pbase + s * 16 + l16;
        pw[s] = pp[s] & (WW - 1);
        ph[s] = (pp[s] >> 7) & (HH - 1);
        pn[s] = pp[s] >> 14;
    }

    const int cos = blockIdx.y;        // 64-out-channel slice
    const bf16_t* wsrc = wpk + (size_t)cos * (KTOT / 32) * 64 * 32;
    const unsigned lds0 = (unsigned)(size_t)(&wlds[0][0]);
    constexpr unsigned LBUF = 64u * KB * 2u;   // bytes per buffer

    v8f acc0 = {0.f, 0.f, 0.f, 0.f, 0.f, 0.f, 0.f, 0.f};
    v8f acc1 = {0.f, 0.f, 0.f, 0.f, 0.f, 0.f, 0.f, 0.f};

    // prologue: stage kb=0
    if (tid < 32) {
        tdm_load_1d(wsrc, lds0, KB * 64 * 2);
        __builtin_amdgcn_s_wait_tensorcnt(0);
    }
    __syncthreads();

    for (int kb = 0; kb < NKB; ++kb) {
        // kick off next stage into the other buffer while we compute
        if (tid < 32 && kb + 1 < NKB) {
            tdm_load_1d(wsrc + (kb + 1) * KB * 64, lds0 + ((kb + 1) & 1) * LBUF, KB * 64 * 2);
            __builtin_prefetch(wsrc + (kb + 1) * KB * 64, 0, 1);
        }
        const bf16_t* cur = &wlds[kb & 1][0];

#pragma unroll
        for (int kc2 = 0; kc2 < KCB; ++kc2) {
            const int kk = (kb * KCB + kc2) * 32;
            const int tap = kk / CIN;
            const int crem = kk - tap * CIN;
            union U { v4u u[2]; v16bf v; } A, B;
            // A fragment: lane=M row, two 16B chunks per the 16-bit A layout
            {
                const bf16_t* ra = cur + (kc2 * 64 + mt * 16 + l16) * 32;
                A.u[0] = *(const v4u*)(ra + lh * 8);
                A.u[1] = *(const v4u*)(ra + 16 + lh * 8);
            }
            int dh = 0, dw = 0;
            if (TAPS == 9) { dh = tap / 3 - 1; dw = tap % 3 - 1; }
#pragma unroll
            for (int s = 0; s < 2; ++s) {
                const int hh = ph[s] + dh, ww = pw[s] + dw;
                bool valid = true;
                if (TAPS == 9) valid = (hh >= 0) && (hh < HH) && (ww >= 0) && (ww < WW);
                const bf16_t* rb = act + (size_t)(((pn[s] * HH + hh) * WW + ww) * CIN
                                                  + crem + lh * 8);
                v4u z = {0u, 0u, 0u, 0u};
                B.u[0] = valid ? *(const v4u*)(rb)     : z;
                B.u[1] = valid ? *(const v4u*)(rb + 16) : z;
                if (s == 0)
                    acc0 = __builtin_amdgcn_wmma_f32_16x16x32_bf16(
                               false, A.v, false, B.v, (short)0, acc0, false, false);
                else
                    acc1 = __builtin_amdgcn_wmma_f32_16x16x32_bf16(
                               false, A.v, false, B.v, (short)0, acc1, false, false);
            }
        }
        if (tid < 32) __builtin_amdgcn_s_wait_tensorcnt(0);
        __syncthreads();
    }

    // Epilogue: D layout -> element r is (M = mt*16 + lh*8 + r, N = l16)
    constexpr int HW = HH * WW;
#pragma unroll
    for (int s = 0; s < 2; ++s) {
#pragma unroll
        for (int r = 0; r < 8; ++r) {
            int mg = cos * 64 + mt * 16 + lh * 8 + r;
            if (mg < COUT) {
                float v = (s == 0 ? acc0[r] : acc1[r]) + bias[mg];
                if (RELU) v = lrelu_f(v);
                if (OUTF == 0)
                    ((bf16_t*)outp)[(size_t)pp[s] * COUT + mg] = (bf16_t)v;
                else
                    ((float*)outp)[(size_t)((pn[s] * COUT + mg) * HW + ph[s] * WW + pw[s])] = v;
            }
        }
    }
}

// ---------------------------------------------------------------------------
// DCN-v2 sampling: offsets = flow(rev) + conv4 residual, mask = sigmoid.
// Writes masked bilinear samples as bf16 [P][tap*64 + g*4 + c]  (K order
// matching the packed dcn weights) for the trailing WMMA contraction.
// ---------------------------------------------------------------------------
__global__ void dcn_sample_kernel(const float* __restrict__ nbr, const bf16_t* __restrict__ out4,
                                  const float* __restrict__ flow, bf16_t* __restrict__ samp) {
    unsigned tid = blockIdx.x * blockDim.x + threadIdx.x;   // < PTOT*144
    int gt = (int)(tid % 144u);
    int p  = (int)(tid / 144u);
    int tap = gt / 16, g = gt % 16;
    int w = p & (WW - 1), h = (p >> 7) & (HH - 1), n = p >> 14;
    constexpr int HW = HH * WW;
    int pix = h * WW + w;

    const bf16_t* o = out4 + (size_t)p * 432;
    int c = g * 9 + tap;
    float fx = flow[(n * 2 + 0) * HW + pix];
    float fy = flow[(n * 2 + 1) * HW + pix];
    float dy = fy + (float)o[2 * c];
    float dx = fx + (float)o[2 * c + 1];
    float mk = (float)o[288 + c];
    mk = 1.f / (1.f + __expf(-mk));

    float py = (float)(h + tap / 3 - 1) + dy;
    float px = (float)(w + tap % 3 - 1) + dx;
    float y0f = floorf(py), x0f = floorf(px);
    float fdy = py - y0f, fdx = px - x0f;
    int y0 = (int)y0f, x0 = (int)x0f, y1 = y0 + 1, x1 = x0 + 1;
    float w00 = (1.f - fdy) * (1.f - fdx) * ((y0 >= 0 && y0 < HH && x0 >= 0 && x0 < WW) ? 1.f : 0.f);
    float w01 = (1.f - fdy) * fdx         * ((y0 >= 0 && y0 < HH && x1 >= 0 && x1 < WW) ? 1.f : 0.f);
    float w10 = fdy * (1.f - fdx)         * ((y1 >= 0 && y1 < HH && x0 >= 0 && x0 < WW) ? 1.f : 0.f);
    float w11 = fdy * fdx                 * ((y1 >= 0 && y1 < HH && x1 >= 0 && x1 < WW) ? 1.f : 0.f);
    int y0c = iclamp(y0, 0, HH - 1), y1c = iclamp(y1, 0, HH - 1);
    int x0c = iclamp(x0, 0, WW - 1), x1c = iclamp(x1, 0, WW - 1);

    bf16_t* dst = samp + (size_t)p * 576 + tap * 64 + g * 4;
#pragma unroll
    for (int cc = 0; cc < 4; ++cc) {
        const float* base = nbr + (n * 64 + g * 4 + cc) * HW;
        float v = base[y0c * WW + x0c] * w00 + base[y0c * WW + x1c] * w01 +
                  base[y1c * WW + x0c] * w10 + base[y1c * WW + x1c] * w11;
        dst[cc] = (bf16_t)(v * mk);
    }
}

// ---------------------------------------------------------------------------
extern "C" void kernel_launch(void* const* d_in, const int* in_sizes, int n_in,
                              void* d_out, int out_size, void* d_ws, size_t ws_size,
                              hipStream_t stream) {
    const long long P = PTOT;

    const float* nbr  = (const float*)d_in[0];
    const float* ref  = (const float*)d_in[1];
    const float* flow = (const float*)d_in[2];
    const float* w1 = (const float*)d_in[3];  const float* b1 = (const float*)d_in[4];
    const float* w2 = (const float*)d_in[5];  const float* b2 = (const float*)d_in[6];
    const float* w3 = (const float*)d_in[7];  const float* b3 = (const float*)d_in[8];
    const float* w4 = (const float*)d_in[9];  const float* b4 = (const float*)d_in[10];
    const float* dw = (const float*)d_in[11]; const float* db = (const float*)d_in[12];

    char* ws = (char*)d_ws;
    auto take = [&](long long bytes) { char* r = ws; ws += (bytes + 255) & ~255ll; return r; };
    bf16_t* x0  = (bf16_t*)take(P * 128 * 2);   // concat(warped, ref)  NHWC
    bf16_t* h1  = (bf16_t*)take(P * 64 * 2);
    bf16_t* h2  = (bf16_t*)take(P * 64 * 2);
    bf16_t* h3  = (bf16_t*)take(P * 64 * 2);
    bf16_t* o4  = (bf16_t*)take(P * 432 * 2);   // offsets/mask raw
    bf16_t* smp = (bf16_t*)take(P * 576 * 2);   // masked deformable samples
    bf16_t* pw1 = (bf16_t*)take(64ll  * 1152 * 2);
    bf16_t* pw2 = (bf16_t*)take(64ll  * 576 * 2);
    bf16_t* pw3 = (bf16_t*)take(64ll  * 576 * 2);
    bf16_t* pw4 = (bf16_t*)take(448ll * 576 * 2);
    bf16_t* pwd = (bf16_t*)take(64ll  * 576 * 2);

    auto blocks = [](long long total) { return dim3((unsigned)((total + 255) / 256)); };

    // 1) repack all weights to WMMA A-fragment order (bf16)
    pack_weights_kernel<<<blocks(64ll * 1152), 256, 0, stream>>>(w1, pw1, 64, 128, 9, 64);
    pack_weights_kernel<<<blocks(64ll * 576),  256, 0, stream>>>(w2, pw2, 64, 64, 9, 64);
    pack_weights_kernel<<<blocks(64ll * 576),  256, 0, stream>>>(w3, pw3, 64, 64, 9, 64);
    pack_weights_kernel<<<blocks(448ll * 576), 256, 0, stream>>>(w4, pw4, 432, 64, 9, 448);
    pack_weights_kernel<<<blocks(64ll * 576),  256, 0, stream>>>(dw, pwd, 64, 64, 9, 64);

    // 2) backwarp + concat -> x0
    warp_concat_kernel<<<blocks(P * 64), 256, 0, stream>>>(nbr, ref, flow, x0);

    // 3) conv stack (implicit GEMM, WMMA bf16, double-buffered TDM weights)
    dim3 cgrid((unsigned)(P / 64), 1);
    conv_wmma_kernel<128, 9, 64, true, 0><<<cgrid, 256, 0, stream>>>(x0, pw1, b1, h1);
    conv_wmma_kernel<64, 9, 64, true, 0><<<cgrid, 256, 0, stream>>>(h1, pw2, b2, h2);
    conv_wmma_kernel<64, 9, 64, true, 0><<<cgrid, 256, 0, stream>>>(h2, pw3, b3, h3);
    dim3 cgrid4((unsigned)(P / 64), 7);          // 432 out-ch -> 7 slices of 64 (padded)
    conv_wmma_kernel<64, 9, 432, false, 0><<<cgrid4, 256, 0, stream>>>(h3, pw4, b4, o4);

    // 4) deformable sampling with flow-guided offsets + sigmoid mask
    dcn_sample_kernel<<<blocks(P * 144), 256, 0, stream>>>(nbr, o4, flow, smp);

    // 5) DCN contraction 576 -> 64 as dense WMMA GEMM, + bias + lrelu, fp32 NCHW out
    conv_wmma_kernel<576, 1, 64, true, 1><<<cgrid, 256, 0, stream>>>(smp, pwd, db, d_out);
}
